// MolecularInteractionModel_46059229282850
// MI455X (gfx1250) — matrix operand order repaired and checked by reading, not compiled, not run on hardware
//
#include <hip/hip_runtime.h>
#include <hip/hip_bf16.h>
#include <math.h>

typedef __attribute__((ext_vector_type(16))) _Float16 v16h;
typedef __attribute__((ext_vector_type(8)))  _Float16 v8h;
typedef __attribute__((ext_vector_type(8)))  float    v8f;

#define DEV_INLINE __device__ __forceinline__

constexpr int Bsz   = 1024;
constexpr int NPG   = 100;
constexpr int NT    = Bsz * NPG;   // 102400 nodes
constexpr int Eg    = NT * 8;      // 819200 edges per side
constexpr int HID   = 128;
constexpr int FEAT  = 4 * HID + 256; // 768

DEV_INLINE float sigm(float x) { return 1.f / (1.f + __expf(-x)); }
// monotonic float <-> uint mapping so atomicMax(uint) == float max
DEV_INLINE unsigned f2ord(float f) {
  unsigned u = __float_as_uint(f);
  return (u & 0x80000000u) ? ~u : (u | 0x80000000u);
}
DEV_INLINE float ord2f(unsigned m) {
  m = (m & 0x80000000u) ? (m & 0x7fffffffu) : ~m;
  return __uint_as_float(m);
}

DEV_INLINE v16h ld16(const _Float16* p0, const _Float16* p1) {
  v8h x = *(const v8h*)p0;
  v8h y = *(const v8h*)p1;
  return __builtin_shufflevector(x, y, 0, 1, 2, 3, 4, 5, 6, 7, 8, 9, 10, 11, 12, 13, 14, 15);
}
#define WMMA(A, B, C) __builtin_amdgcn_wmma_f32_16x16x32_f16(false, A, false, B, (short)0, C, false, false)

// ---------------------------------------------------------------------------
// f32 [R,C] -> zero-padded f16 [Rp,Cp]   (n = Rp*Cp threads)
// ---------------------------------------------------------------------------
__global__ void k_cvt_f16(const float* __restrict__ src, _Float16* __restrict__ dst,
                          int R, int C, int Cp, int n) {
  int i = blockIdx.x * blockDim.x + threadIdx.x;
  if (i >= n) return;
  int r = i / Cp, c = i - r * Cp;
  dst[i] = (r < R && c < C) ? (_Float16)src[(long)r * C + c] : (_Float16)0.f;
}

// ---------------------------------------------------------------------------
// GEMM: C[M,Nc] (+)= act( A16[M,Kp] * W16[Ncp,Kp]^T + bias )
// 32x64 output per wave32 (2 row-tiles x 4 col-tiles), 8 v_wmma per K-step.
// Register double-buffered: next K-step's 12 x b128 loads issue before the
// current step's WMMAs, so s_wait_loadcnt lands a full step away from use.
// Operand slabs are padded +64 halves so the tail over-read is harmless.
// Requires M % 32 == 0, Kp % 32 == 0, Ncp % 64 == 0.
// ---------------------------------------------------------------------------
__global__ void __launch_bounds__(256)
k_gemm_wmma(const _Float16* __restrict__ A16, const _Float16* __restrict__ W16,
            const float* __restrict__ bias, float* __restrict__ C,
            int M, int Nc, int Kp, int Ncp, int accumulate, int relu) {
  const int lane = threadIdx.x & 31;
  const int l15  = lane & 15;
  const int hi   = lane >> 4;
  const int ngx  = Ncp >> 6;
  const int grp  = blockIdx.x * (blockDim.x >> 5) + (threadIdx.x >> 5);
  if (grp >= ngx * (M >> 5)) return;             // wave-uniform: EXEC stays full
  const int row0 = (grp / ngx) << 5;
  const int col0 = (grp % ngx) << 6;

  v8f acc[2][4];
#pragma unroll
  for (int i = 0; i < 2; ++i)
#pragma unroll
    for (int j = 0; j < 4; ++j) {
      if (accumulate) {
#pragma unroll
        for (int r = 0; r < 8; ++r) {
          int row = row0 + i * 16 + hi * 8 + r;
          int col = col0 + j * 16 + l15;
          acc[i][j][r] = (col < Nc) ? C[(long)row * Nc + col] : 0.f;
        }
      } else {
        acc[i][j] = (v8f){0.f, 0.f, 0.f, 0.f, 0.f, 0.f, 0.f, 0.f};
      }
    }

  // A fragment: row = row0 + i*16 + l15 ; halves k0 + hi*16 + 0..15 (contiguous)
  const _Float16* Ar0 = A16 + (size_t)(row0 + l15) * Kp + hi * 16;
  const _Float16* Ar1 = Ar0 + (size_t)16 * Kp;
  // B fragment: col = col0 + j*16 + l15 ; halves k0 + hi*16 + 0..15 (contiguous)
  const _Float16* Bc0 = W16 + (size_t)(col0 + l15) * Kp + hi * 16;
  const _Float16* Bc1 = Bc0 + (size_t)16 * Kp;
  const _Float16* Bc2 = Bc0 + (size_t)32 * Kp;
  const _Float16* Bc3 = Bc0 + (size_t)48 * Kp;

  // prologue loads (k0 = 0)
  v16h a0 = ld16(Ar0, Ar0 + 8), a1 = ld16(Ar1, Ar1 + 8);
  v16h b0 = ld16(Bc0, Bc0 + 8),  b1 = ld16(Bc1, Bc1 + 8);
  v16h b2 = ld16(Bc2, Bc2 + 8),  b3 = ld16(Bc3, Bc3 + 8);

  for (int k0 = 0; k0 < Kp; k0 += 32) {
    const int kn = k0 + 32;                      // over-reads padded tail on last iter
    v16h a0n = ld16(Ar0 + kn, Ar0 + kn + 8);
    v16h a1n = ld16(Ar1 + kn, Ar1 + kn + 8);
    v16h b0n = ld16(Bc0 + kn, Bc0 + kn + 8);
    v16h b1n = ld16(Bc1 + kn, Bc1 + kn + 8);
    v16h b2n = ld16(Bc2 + kn, Bc2 + kn + 8);
    v16h b3n = ld16(Bc3 + kn, Bc3 + kn + 8);
    __builtin_prefetch(Ar0 + k0 + 320, 0, 0);    // global_prefetch_b8, ~10 steps ahead
    __builtin_prefetch(Ar1 + k0 + 320, 0, 0);
    __builtin_prefetch(Bc0 + k0 + 320, 0, 0);
    __builtin_prefetch(Bc1 + k0 + 320, 0, 0);
    __builtin_prefetch(Bc2 + k0 + 320, 0, 0);
    __builtin_prefetch(Bc3 + k0 + 320, 0, 0);
    acc[0][0] = WMMA(a0, b0, acc[0][0]);
    acc[1][0] = WMMA(a1, b0, acc[1][0]);
    acc[0][1] = WMMA(a0, b1, acc[0][1]);
    acc[1][1] = WMMA(a1, b1, acc[1][1]);
    acc[0][2] = WMMA(a0, b2, acc[0][2]);
    acc[1][2] = WMMA(a1, b2, acc[1][2]);
    acc[0][3] = WMMA(a0, b3, acc[0][3]);
    acc[1][3] = WMMA(a1, b3, acc[1][3]);
    a0 = a0n; a1 = a1n; b0 = b0n; b1 = b1n; b2 = b2n; b3 = b3n;
  }

#pragma unroll
  for (int i = 0; i < 2; ++i)
#pragma unroll
    for (int j = 0; j < 4; ++j) {
      int col = col0 + j * 16 + l15;
      if (col < Nc) {
        float bv = bias ? bias[col] : 0.f;
#pragma unroll
        for (int r = 0; r < 8; ++r) {
          int row = row0 + i * 16 + hi * 8 + r;
          float v = acc[i][j][r] + bv;
          if (relu) v = fmaxf(v, 0.f);
          C[(long)row * Nc + col] = v;
        }
      }
    }
}

// ---------------------------------------------------------------------------
// Elementwise / reduction kernels
// ---------------------------------------------------------------------------
__global__ void k_fill_f32(float* p, float v, int n) {
  int i = blockIdx.x * blockDim.x + threadIdx.x;
  if (i < n) p[i] = v;
}
__global__ void k_fill_u32(unsigned* p, unsigned v, int n) {
  int i = blockIdx.x * blockDim.x + threadIdx.x;
  if (i < n) p[i] = v;
}

__global__ void k_l2norm(const float* __restrict__ x, float* __restrict__ y, int D) {
  __shared__ float red[256];
  int b = blockIdx.x;
  float s = 0.f;
  for (int d = threadIdx.x; d < D; d += 256) { float v = x[(long)b * D + d]; s += v * v; }
  red[threadIdx.x] = s; __syncthreads();
  for (int o = 128; o > 0; o >>= 1) { if (threadIdx.x < o) red[threadIdx.x] += red[threadIdx.x + o]; __syncthreads(); }
  float inv = 1.f / fmaxf(sqrtf(red[0]), 1e-12f);
  for (int d = threadIdx.x; d < D; d += 256) y[(long)b * D + d] = x[(long)b * D + d] * inv;
}

__global__ void k_add_ec(const float* __restrict__ x, const float* __restrict__ ec,
                         float* __restrict__ out, int n) { // n = NT*78
  int i = blockIdx.x * blockDim.x + threadIdx.x;
  if (i >= n) return;
  int node = i / 78, j = i - node * 78;
  out[i] = x[i] + ec[(node / NPG) * 78 + j];
}

__global__ void k_gat_coef(const float* __restrict__ xp, const float* __restrict__ as_,
                           const float* __restrict__ ad_, float* __restrict__ asrc,
                           float* __restrict__ adst) {
  int i = blockIdx.x * blockDim.x + threadIdx.x;   // i = n*4+h
  if (i >= NT * 4) return;
  int h = i & 3, n = i >> 2;
  float s = 0.f, d0 = 0.f;
#pragma unroll 8
  for (int d = 0; d < 32; ++d) {
    float v = xp[(long)n * HID + h * 32 + d];
    s  += v * as_[h * 32 + d];
    d0 += v * ad_[h * 32 + d];
  }
  asrc[i] = s; adst[i] = d0;
}

__global__ void k_edge_alpha_max(const int* __restrict__ ei, const float* __restrict__ asrc,
                                 const float* __restrict__ adst, float* __restrict__ alpha,
                                 unsigned* __restrict__ mx) {
  int i = blockIdx.x * blockDim.x + threadIdx.x;   // (edge, head)
  const int EE = Eg + NT;
  if (i >= EE * 4) return;
  int h = i & 3, e = i >> 2;
  int s = (e < Eg) ? ei[e] : (e - Eg);
  int d = (e < Eg) ? ei[Eg + e] : (e - Eg);
  float a = asrc[s * 4 + h] + adst[d * 4 + h];
  a = (a > 0.f) ? a : 0.2f * a;                    // leaky_relu 0.2
  alpha[i] = a;
  atomicMax(&mx[d * 4 + h], f2ord(a));
}

__global__ void k_edge_expsum(const int* __restrict__ ei, float* __restrict__ alpha,
                              const unsigned* __restrict__ mx, float* __restrict__ den) {
  int i = blockIdx.x * blockDim.x + threadIdx.x;
  const int EE = Eg + NT;
  if (i >= EE * 4) return;
  int h = i & 3, e = i >> 2;
  int d = (e < Eg) ? ei[Eg + e] : (e - Eg);
  float ex = __expf(alpha[i] - ord2f(mx[d * 4 + h]));
  alpha[i] = ex;
  atomicAdd(&den[d * 4 + h], ex);
}

__global__ void k_edge_agg(const int* __restrict__ ei, const float* __restrict__ alpha,
                           const float* __restrict__ den, const float* __restrict__ xp,
                           float* __restrict__ out) {
  int i = blockIdx.x * blockDim.x + threadIdx.x;
  const int EE = Eg + NT;
  if (i >= EE * 4) return;
  int h = i & 3, e = i >> 2;
  int s = (e < Eg) ? ei[e] : (e - Eg);
  int d = (e < Eg) ? ei[Eg + e] : (e - Eg);
  float w = alpha[i] / (den[d * 4 + h] + 1e-16f);
  const float* xs = xp + (long)s * HID + h * 32;
  float* od = out + (long)d * HID + h * 32;
#pragma unroll
  for (int t = 0; t < 32; ++t) atomicAdd(&od[t], xs[t] * w);
}

__global__ void k_bias_add(float* __restrict__ out, const float* __restrict__ b, int n) {
  int i = blockIdx.x * blockDim.x + threadIdx.x;
  if (i < n) out[i] += b[i & (HID - 1)];
}

__global__ void k_lstm_cell(const float* __restrict__ gates, float* __restrict__ h,
                            float* __restrict__ cst) {
  int i = blockIdx.x * blockDim.x + threadIdx.x;   // n*128+d
  if (i >= NT * HID) return;
  int n = i >> 7, d = i & 127;
  const float* g = gates + (long)n * 512;
  float c2 = sigm(g[128 + d]) * cst[i] + sigm(g[d]) * tanhf(g[256 + d]);
  float h2 = sigm(g[384 + d]) * tanhf(c2);
  cst[i] = c2; h[i] = h2;
}

// s[b,k] = sum_q tanh( dot(kk[b,k,:], qq[b,q,:]) )   -- one block per (b,k)
__global__ void k_attn_scores(const float* __restrict__ kk, const float* __restrict__ qq,
                              float* __restrict__ s) {
  __shared__ float kv[HID];
  __shared__ float red[HID];
  int bk = blockIdx.x, b = bk / NPG, t = threadIdx.x;
  kv[t] = kk[(long)bk * HID + t];
  __syncthreads();
  float acc = 0.f;
  if (t < NPG) {
    const float* qp = qq + ((long)b * NPG + t) * HID;
    float dot = 0.f;
    for (int d = 0; d < HID; ++d) dot += kv[d] * qp[d];
    acc = tanhf(dot);
  }
  red[t] = acc; __syncthreads();
  for (int o = 64; o > 0; o >>= 1) { if (t < o) red[t] += red[t + o]; __syncthreads(); }
  if (t == 0) s[bk] = red[0];
}

__global__ void k_masked_softmax(const float* __restrict__ s, const int* __restrict__ mask,
                                 float* __restrict__ w) {
  __shared__ float red[HID];
  int b = blockIdx.x, t = threadIdx.x;
  bool valid = (t < NPG) && (mask[b * NPG + t] == 0);
  float v = valid ? s[b * NPG + t] : -1e30f;
  red[t] = v; __syncthreads();
  for (int o = 64; o > 0; o >>= 1) { if (t < o) red[t] = fmaxf(red[t], red[t + o]); __syncthreads(); }
  float m = red[0]; __syncthreads();
  float e = valid ? __expf(s[b * NPG + t] - m) : 0.f;
  red[t] = e; __syncthreads();
  for (int o = 64; o > 0; o >>= 1) { if (t < o) red[t] += red[t + o]; __syncthreads(); }
  float inv = 1.f / red[0];
  if (t < NPG) w[b * NPG + t] = e * inv;
}

// o[b,:] = layernorm(mean_k f[b,k,:] + sum_k w[b,k]*v[b,k,:]) * g + be -> feat[:, off:off+128]
__global__ void k_attn_out(const float* __restrict__ f, const float* __restrict__ v,
                           const float* __restrict__ w, const float* __restrict__ g,
                           const float* __restrict__ be, float* __restrict__ feat, int off) {
  __shared__ float wsh[NPG];
  __shared__ float red[HID];
  int b = blockIdx.x, t = threadIdx.x;
  if (t < NPG) wsh[t] = w[b * NPG + t];
  __syncthreads();
  float cacc = 0.f, macc = 0.f;
  for (int k = 0; k < NPG; ++k) {
    long idx = ((long)b * NPG + k) * HID + t;
    cacc += wsh[k] * v[idx];
    macc += f[idx];
  }
  float x = macc * (1.f / NPG) + cacc;
  red[t] = x; __syncthreads();
  for (int o = 64; o > 0; o >>= 1) { if (t < o) red[t] += red[t + o]; __syncthreads(); }
  float mu = red[0] * (1.f / HID); __syncthreads();
  float dv = x - mu;
  red[t] = dv * dv; __syncthreads();
  for (int o = 64; o > 0; o >>= 1) { if (t < o) red[t] += red[t + o]; __syncthreads(); }
  float var = red[0] * (1.f / HID);
  feat[(long)b * FEAT + off + t] = dv * rsqrtf(var + 1e-5f) * g[t] + be[t];
}

__global__ void k_copy_tc(const float* __restrict__ tc, float* __restrict__ feat) {
  int i = blockIdx.x * blockDim.x + threadIdx.x;   // B*256
  if (i >= Bsz * 256) return;
  feat[(long)(i >> 8) * FEAT + 512 + (i & 255)] = tc[i];
}

// ---------------------------------------------------------------------------
extern "C" void kernel_launch(void* const* d_in, const int* in_sizes, int n_in,
                              void* d_out, int out_size, void* d_ws, size_t ws_size,
                              hipStream_t stream) {
  (void)in_sizes; (void)n_in; (void)out_size; (void)ws_size;
  const float* left_x  = (const float*)d_in[0];
  const float* right_x = (const float*)d_in[1];
  const float* cell    = (const float*)d_in[2];
  const int*   lei     = (const int*)d_in[3];
  const int*   rei     = (const int*)d_in[4];
  /* d_in[5] = batch : node->graph is n/NPG by construction */
  const int*   lmask   = (const int*)d_in[6];
  const int*   rmask   = (const int*)d_in[7];
  const float* P[64];
  for (int i = 8; i < 52; ++i) P[i] = (const float*)d_in[i];
  // 8..11 gat0{W,as,ad,b}  12..15 gat1  16..19 lstm{Wih,Whh,bih,bhh}
  // 20..25 ft1(W,b x3)  26..31 ft2  32..39 seq_attn  40..47 graph_attn  48..51 head

  char* base = (char*)d_ws;
  size_t off = 0;
  auto allocB = [&](size_t bytes) -> void* {
    void* p = (void*)(base + off);
    off += (bytes + 255) & ~(size_t)255;
    return p;
  };
  auto alloc  = [&](size_t nfl) -> float*    { return (float*)allocB(nfl * 4); };
  auto allocH = [&](size_t nh)  -> _Float16* { return (_Float16*)allocB((nh + 64) * 2); }; // +64 halves tail pad

  float* cn    = alloc((size_t)Bsz * 954);
  float* t2048 = alloc((size_t)Bsz * 2048);
  float* t512  = alloc((size_t)Bsz * 512);
  float* ec    = alloc((size_t)Bsz * 78);
  float* tc    = alloc((size_t)Bsz * 256);
  float* feat  = alloc((size_t)Bsz * FEAT);
  float* h64   = alloc((size_t)Bsz * 64);
  float* s1    = alloc((size_t)Bsz * NPG);
  float* s2    = alloc((size_t)Bsz * NPG);
  float* w1    = alloc((size_t)Bsz * NPG);
  float* w2    = alloc((size_t)Bsz * NPG);
  const size_t NH = (size_t)NT * HID;
  float* lg = alloc(NH); float* rg = alloc(NH);
  float* lh = alloc(NH); float* lc = alloc(NH);
  float* rh = alloc(NH); float* rc = alloc(NH);
  float* xp    = alloc(NH);
  float* gates = alloc((size_t)NT * 512);
  float* asrc  = alloc((size_t)NT * 4);
  float* adst  = alloc((size_t)NT * 4);
  float* alpha = alloc((size_t)(Eg + NT) * 4);
  unsigned* mxu = (unsigned*)alloc((size_t)NT * 4);
  float* den   = alloc((size_t)NT * 4);
  _Float16* A16 = allocH((size_t)NT * 128);   // largest activation: [NT,128]

  auto cdiv = [](int a, int b) { return (a + b - 1) / b; };
  auto rK = [](int k) { return (k + 31) & ~31; };
  auto rN = [](int n) { return (n + 63) & ~63; };

  // ---- convert all weights to padded f16 once ----
  auto cvtW = [&](const float* W, int Nc, int K) -> _Float16* {
    int Kp = rK(K), Ncp = rN(Nc);
    _Float16* p = allocH((size_t)Ncp * Kp);
    int n = Ncp * Kp;
    k_cvt_f16<<<cdiv(n, 256), 256, 0, stream>>>(W, p, Nc, K, Kp, n);
    return p;
  };
  _Float16* w_gat[2] = { cvtW(P[8], HID, 78), cvtW(P[12], HID, HID) };
  _Float16* w_ih  = cvtW(P[16], 512, HID);
  _Float16* w_hh  = cvtW(P[17], 512, HID);
  _Float16* w_f1a = cvtW(P[20], 2048, 954);
  _Float16* w_f1b = cvtW(P[22], 512, 2048);
  _Float16* w_f1c = cvtW(P[24], 256, 512);
  _Float16* w_f2a = cvtW(P[26], 2048, 954);
  _Float16* w_f2b = cvtW(P[28], 512, 2048);
  _Float16* w_f2c = cvtW(P[30], 78, 512);
  _Float16* w_att[6] = { cvtW(P[32], HID, HID), cvtW(P[34], HID, HID), cvtW(P[36], HID, HID),
                         cvtW(P[40], HID, HID), cvtW(P[42], HID, HID), cvtW(P[44], HID, HID) };
  _Float16* w_h0 = cvtW(P[48], 64, FEAT);
  _Float16* w_h1 = cvtW(P[50], 2, 64);

  // ---- activation conversion with memoization ----
  const float* curA = nullptr; int curK = -1;
  auto cvtA = [&](const float* Asrc, int M, int K) -> int {
    int Kp = rK(K);
    if (curA != Asrc || curK != K) {
      curA = Asrc; curK = K;
      int n = M * Kp;
      k_cvt_f16<<<cdiv(n, 256), 256, 0, stream>>>(Asrc, A16, M, K, Kp, n);
    }
    return Kp;
  };
  auto gemm = [&](const float* Asrc, int K, const _Float16* W16, const float* bias,
                  float* C, int M, int Nc, int acc, int relu) {
    int Kp = cvtA(Asrc, M, K);
    int Ncp = rN(Nc);
    int groups = (M >> 5) * (Ncp >> 6);
    k_gemm_wmma<<<cdiv(groups, 8), 256, 0, stream>>>(A16, W16, bias, C, M, Nc, Kp, Ncp, acc, relu);
  };
  auto fillf = [&](float* p, float v, int n) { k_fill_f32<<<cdiv(n, 256), 256, 0, stream>>>(p, v, n); };
  auto fillu = [&](unsigned* p, unsigned v, int n) { k_fill_u32<<<cdiv(n, 256), 256, 0, stream>>>(p, v, n); };

  // 1) cn = normalize(cell); ft2 -> ec ; ft1 -> tc
  k_l2norm<<<Bsz, 256, 0, stream>>>(cell, cn, 954);
  gemm(cn, 954, w_f2a, P[27], t2048, Bsz, 2048, 0, 1);
  gemm(t2048, 2048, w_f2b, P[29], t512, Bsz, 512, 0, 1);
  gemm(t512, 512, w_f2c, P[31], ec, Bsz, 78, 0, 1);
  gemm(cn, 954, w_f1a, P[21], t2048, Bsz, 2048, 0, 1);
  gemm(t2048, 2048, w_f1b, P[23], t512, Bsz, 512, 0, 1);
  gemm(t512, 512, w_f1c, P[25], tc, Bsz, 256, 0, 1);

  // 2) lg/rg = x + exp_cell ; zero LSTM state
  k_add_ec<<<cdiv(NT * 78, 256), 256, 0, stream>>>(left_x, ec, lg, NT * 78);
  k_add_ec<<<cdiv(NT * 78, 256), 256, 0, stream>>>(right_x, ec, rg, NT * 78);
  fillf(lh, 0.f, NT * HID); fillf(lc, 0.f, NT * HID);
  fillf(rh, 0.f, NT * HID); fillf(rc, 0.f, NT * HID);

  // 3) two GAT + LSTM layers per side
  const int EE4 = (Eg + NT) * 4;
  const int gatK[2] = { 78, HID };
  const float* gatAS[2] = { P[9],  P[13] };
  const float* gatAD[2] = { P[10], P[14] };
  const float* gatB[2]  = { P[11], P[15] };
  for (int layer = 0; layer < 2; ++layer) {
    struct { float *x, *h, *c; const int* ei; } sides[2] = {
      { lg, lh, lc, lei }, { rg, rh, rc, rei } };
    for (int s = 0; s < 2; ++s) {
      float* x = sides[s].x; float* h = sides[s].h; float* c = sides[s].c;
      const int* ei = sides[s].ei;
      gemm(x, gatK[layer], w_gat[layer], nullptr, xp, NT, HID, 0, 0);     // xp = x @ W.T
      k_gat_coef<<<cdiv(NT * 4, 256), 256, 0, stream>>>(xp, gatAS[layer], gatAD[layer], asrc, adst);
      fillu(mxu, 0u, NT * 4);
      fillf(den, 0.f, NT * 4);
      fillf(x, 0.f, NT * HID);                                            // x reused as aggregate output
      k_edge_alpha_max<<<cdiv(EE4, 256), 256, 0, stream>>>(ei, asrc, adst, alpha, mxu);
      k_edge_expsum<<<cdiv(EE4, 256), 256, 0, stream>>>(ei, alpha, mxu, den);
      k_edge_agg<<<cdiv(EE4, 256), 256, 0, stream>>>(ei, alpha, den, xp, x);
      k_bias_add<<<cdiv(NT * HID, 256), 256, 0, stream>>>(x, gatB[layer], NT * HID);
      // LSTM: gates = x@Wih.T + bih ; gates += h@Whh.T + bhh ; update (h,c)
      gemm(x, HID, w_ih, P[18], gates, NT, 512, 0, 0);
      gemm(h, HID, w_hh, P[19], gates, NT, 512, 1, 0);
      k_lstm_cell<<<cdiv(NT * HID, 256), 256, 0, stream>>>(gates, h, c);
    }
  }

  // 4) dual attention (seq uses final lh/rh == lseq/rseq; graph uses lg/rg)
  //    qkv scratch: gates holds 4 buffers; dead lc/rc hold the other 2
  float* q1 = gates;           float* k1 = gates + NH;
  float* v1 = gates + 2 * NH;  float* q2 = gates + 3 * NH;
  float* k2 = rc;              float* v2 = lc;
  auto dual_attn = [&](const float* f1, const float* f2, int wb, int pb, int off1, int off2) {
    gemm(f1, HID, w_att[wb + 0], P[pb + 1], q1, NT, HID, 0, 1);
    gemm(f1, HID, w_att[wb + 1], P[pb + 3], k1, NT, HID, 0, 1);
    gemm(f1, HID, w_att[wb + 2], P[pb + 5], v1, NT, HID, 0, 1);
    gemm(f2, HID, w_att[wb + 0], P[pb + 1], q2, NT, HID, 0, 1);
    gemm(f2, HID, w_att[wb + 1], P[pb + 3], k2, NT, HID, 0, 1);
    gemm(f2, HID, w_att[wb + 2], P[pb + 5], v2, NT, HID, 0, 1);
    k_attn_scores<<<NT, HID, 0, stream>>>(k1, q2, s1);
    k_attn_scores<<<NT, HID, 0, stream>>>(k2, q1, s2);
    k_masked_softmax<<<Bsz, HID, 0, stream>>>(s1, lmask, w1);
    k_masked_softmax<<<Bsz, HID, 0, stream>>>(s2, rmask, w2);
    k_attn_out<<<Bsz, HID, 0, stream>>>(f1, v1, w1, P[pb + 6], P[pb + 7], feat, off1);
    k_attn_out<<<Bsz, HID, 0, stream>>>(f2, v2, w2, P[pb + 6], P[pb + 7], feat, off2);
  };
  dual_attn(lh, rh, 0, 32, 256, 384);   // ps1, ps2 (seq_attn)
  dual_attn(lg, rg, 3, 40, 0, 128);     // pg1, pg2 (graph_attn)
  k_copy_tc<<<cdiv(Bsz * 256, 256), 256, 0, stream>>>(tc, feat);

  // 5) head: [B,768] -> relu [B,64] -> [B,2]
  gemm(feat, FEAT, w_h0, P[49], h64, Bsz, 64, 0, 1);
  gemm(h64, 64, w_h1, P[51], (float*)d_out, Bsz, 2, 0, 0);
}
